// Generator_16776142258318
// MI455X (gfx1250) — matrix operand order repaired and checked by reading, not compiled, not run on hardware
//
#include <hip/hip_runtime.h>
#include <hip/hip_bf16.h>

typedef __attribute__((ext_vector_type(16))) _Float16 v16h;
typedef __attribute__((ext_vector_type(8)))  _Float16 v8h;
typedef __attribute__((ext_vector_type(8)))  float    v8f;

#define BATCH   256
#define LPTS    20
#define DIN     3
#define HID     256
#define NCLS    10
#define KPAD    288          // 256 (z) + 3 (xt) padded up to 9 k-blocks of 32
#define NSTEPS  380          // (L-1)/0.05
#define BQ      32           // batch rows per workgroup
#define NWG     (BATCH/BQ)   // 8 workgroups
#define NT      4            // 16x16 output tiles per wave (independent WMMA chains)
#define WSZ     (HID*KPAD)   // one padded weight matrix, in halves

// ---------------------------------------------------------------------------
// Weight prep: convert+pad 5 matrices (W_in, W_h0, W_h1, W_out, Wc1) to f16
// row-major [256][288], zero-padded beyond the real K extent.
// ---------------------------------------------------------------------------
__global__ void prep_weights(const float* __restrict__ W_in,
                             const float* __restrict__ W_h,
                             const float* __restrict__ W_out,
                             const float* __restrict__ Wc1,
                             _Float16* __restrict__ Wf)
{
    int idx = blockIdx.x * 256 + threadIdx.x;
    if (idx >= 5 * WSZ) return;
    int l   = idx / WSZ;
    int rem = idx - l * WSZ;
    int n   = rem / KPAD;
    int k   = rem - n * KPAD;
    float v = 0.f;
    if (l == 0)      { if (k < HID + DIN) v = W_in[n * (HID + DIN) + k]; }
    else if (l <= 2) { if (k < HID)       v = W_h[(l - 1) * HID * HID + n * HID + k]; }
    else if (l == 3) { if (k < HID)       v = W_out[n * HID + k]; }
    else             { if (k < HID)       v = Wc1[n * HID + k]; }
    Wf[idx] = (_Float16)v;
}

// ---------------------------------------------------------------------------
// y0 = X(times[0]) @ W_init.T + b_init  (X(t0) == coeffs[:,0,:])
// ---------------------------------------------------------------------------
__global__ void init_y0(const float* __restrict__ coeffs,
                        const float* __restrict__ W_init,
                        const float* __restrict__ b_init,
                        float* __restrict__ y0)
{
    int idx = blockIdx.x * 256 + threadIdx.x;   // BATCH*HID total
    int m = idx >> 8, n = idx & 255;
    float s = b_init[n];
#pragma unroll
    for (int d = 0; d < DIN; ++d)
        s += coeffs[m * LPTS * DIN + d] * W_init[n * DIN + d];
    y0[idx] = s;
}

// ---------------------------------------------------------------------------
// One [16x(9*32)] x [(9*32)x(NT*16)] GEMM strip per wave using
// v_wmma_f32_16x16x32_f16.  A: LDS activations [BQ][288] f16 (row-major).
// W: global f16 [256][288].  acc[j] = 16x16 tile at columns (cb0+j)*16.
// ---------------------------------------------------------------------------
__device__ __forceinline__ void gemm_layer(const _Float16* __restrict__ W,
                                           const _Float16* __restrict__ A,
                                           int m_base, int cb0, int lane,
                                           v8f acc[NT])
{
    const int klo8  = (lane >> 4) << 3;   // unused: A-frag also starts lanes 16-31 at K+16
    const int klo16 = (lane >> 4) << 4;   // B-frag: lanes 16-31 start at K+16
    const int row   = m_base + (lane & 15);
    const int nl    = lane & 15;
    for (int kb = 0; kb < KPAD / 32; ++kb) {
        const int kbase = kb * 32;
        union { v16h v; v8h h[2]; } af;
        const _Float16* pa = A + row * KPAD + kbase + klo16;
        af.h[0] = *(const v8h*)(pa);          // K chunk [klo16, klo16+8)
        af.h[1] = *(const v8h*)(pa + 8);      // K chunk [klo16+8, klo16+16)
#pragma unroll
        for (int j = 0; j < NT; ++j) {
            const _Float16* pb = W + ((cb0 + j) * 16 + nl) * KPAD + kbase + klo16;
            v16h bf = *(const v16h*)pb;       // 16 contiguous K of weight row n
            acc[j] = __builtin_amdgcn_wmma_f32_16x16x32_f16(
                         false, af.v, false, bf, (short)0, acc[j], false, false);
        }
    }
}

// ---------------------------------------------------------------------------
// Persistent CDE kernel: 8 WGs x 256 threads (8 wave32).  Each WG owns 32
// batch rows; z stays in registers (16x64 f32 per wave) for all 380 steps.
// Wave mapping: row block rb = wave>>2, col blocks cb0..cb0+3, cb0=(wave&3)*4.
// ---------------------------------------------------------------------------
__global__ void __launch_bounds__(256, 1)
cde_kernel(const float* __restrict__ coeffs,
           const float* __restrict__ times,
           const _Float16* __restrict__ Wf,     // [5][256*288] f16
           const float* __restrict__ b_in,
           const float* __restrict__ b_h,       // [2][256]
           const float* __restrict__ b_out,
           const float* __restrict__ bc1,
           const float* __restrict__ Wc2,       // [10][256]
           const float* __restrict__ bc2,
           const float* __restrict__ y0,
           float* __restrict__ out)
{
    __shared__ __align__(64) _Float16 Abuf[2][BQ * KPAD];
    __shared__ float timesS[32];

    const int tid  = threadIdx.x;
    const int wg   = blockIdx.x;
    const int wave = tid >> 5;
    const int lane = tid & 31;
    const int rb     = wave >> 2;          // row block 0..1
    const int cb0    = (wave & 3) * NT;    // first of NT col blocks
    const int m_base = rb * 16;
    const int nl     = lane & 15;
    const int hi8    = (lane >> 4) * 8;    // C-layout row offset for lanes 16-31

    if (tid < LPTS) timesS[tid] = times[tid];
    // Zero the pad columns [256,288) of both buffers once; never touched again
    for (int idx = tid; idx < BQ * 32; idx += 256) {
        int r = idx >> 5, c = idx & 31;
        Abuf[0][r * KPAD + 256 + c] = (_Float16)0.f;
        Abuf[1][r * KPAD + 256 + c] = (_Float16)0.f;
    }
    __syncthreads();

    const float tstart = timesS[0];
    const float dt = (timesS[LPTS - 1] - tstart) / (float)NSTEPS;

    // Load this wave's z tile (16 rows x NT*16 cols) into registers
    v8f z[NT] = {};
#pragma unroll
    for (int j = 0; j < NT; ++j)
#pragma unroll
        for (int i = 0; i < 8; ++i) {
            int m = wg * BQ + m_base + i + hi8;
            int n = (cb0 + j) * 16 + nl;
            z[j][i] = y0[m * HID + n];
        }

    float t = tstart;
    for (int step = 0; step < NSTEPS; ++step) {
        // ---- stage layer-0 input: z (f16) into Abuf[0] cols [0,256) ----
#pragma unroll
        for (int j = 0; j < NT; ++j) {
            int n = (cb0 + j) * 16 + nl;
#pragma unroll
            for (int i = 0; i < 8; ++i)
                Abuf[0][(m_base + i + hi8) * KPAD + n] = (_Float16)z[j][i];
        }
        // ---- xt = linear interp of control path, cols [256,259) ----
        if (tid < BQ) {
            int r = tid, m = wg * BQ + r;
            int idx = 0;
#pragma unroll
            for (int k = 1; k <= LPTS - 2; ++k) idx = (t >= timesS[k]) ? k : idx;
            float ta = timesS[idx], tb = timesS[idx + 1];
            float w = (t - ta) / (tb - ta);
#pragma unroll
            for (int d = 0; d < DIN; ++d) {
                float x0 = coeffs[(m * LPTS + idx) * DIN + d];
                float x1 = coeffs[(m * LPTS + idx + 1) * DIN + d];
                Abuf[0][r * KPAD + 256 + d] = (_Float16)(x0 + (x1 - x0) * w);
            }
        }
        __syncthreads();

        // ---- vector field MLP: 3x lipswish layers + tanh output layer ----
        int cur = 0;
        for (int l = 0; l < 4; ++l) {
            const _Float16* W = Wf + (size_t)l * WSZ;
            if (l < 3) __builtin_prefetch(Wf + (size_t)(l + 1) * WSZ, 0, 1);
            const float* bias = (l == 0) ? b_in : (l == 1) ? b_h
                              : (l == 2) ? (b_h + HID) : b_out;
            v8f acc[NT] = {};
            gemm_layer(W, Abuf[cur], m_base, cb0, lane, acc);
            if (l < 3) {
                int nxt = cur ^ 1;
#pragma unroll
                for (int j = 0; j < NT; ++j) {
                    int n = (cb0 + j) * 16 + nl;
                    float bv = bias[n];
#pragma unroll
                    for (int i = 0; i < 8; ++i) {
                        float x = acc[j][i] + bv;
                        float h = 0.909f * x / (1.f + __expf(-x));   // lipswish
                        Abuf[nxt][(m_base + i + hi8) * KPAD + n] = (_Float16)h;
                    }
                }
                cur = nxt;
            } else {
#pragma unroll
                for (int j = 0; j < NT; ++j) {
                    int n = (cb0 + j) * 16 + nl;
                    float bv = bias[n];
#pragma unroll
                    for (int i = 0; i < 8; ++i) {
                        float x = acc[j][i] + bv;
                        z[j][i] += dt * tanhf(x);                    // Euler step
                    }
                }
            }
            __syncthreads();
        }
        t += dt;
    }

    // ---- classifier: relu(z @ Wc1.T + bc1) @ Wc2.T + bc2 ----
#pragma unroll
    for (int j = 0; j < NT; ++j) {
        int n = (cb0 + j) * 16 + nl;
#pragma unroll
        for (int i = 0; i < 8; ++i)
            Abuf[0][(m_base + i + hi8) * KPAD + n] = (_Float16)z[j][i];
    }
    __syncthreads();
    {
        v8f acc[NT] = {};
        gemm_layer(Wf + (size_t)4 * WSZ, Abuf[0], m_base, cb0, lane, acc);
#pragma unroll
        for (int j = 0; j < NT; ++j) {
            int n = (cb0 + j) * 16 + nl;
            float bv = bc1[n];
#pragma unroll
            for (int i = 0; i < 8; ++i) {
                float x = acc[j][i] + bv;
                Abuf[1][(m_base + i + hi8) * KPAD + n] = (_Float16)fmaxf(x, 0.f);
            }
        }
    }
    __syncthreads();
    // tiny final GEMM [32,256]x[256,10] done scalar
    for (int idx = tid; idx < BQ * NCLS; idx += 256) {
        int m = idx / NCLS, c = idx - m * NCLS;
        float s = bc2[c];
        for (int k = 0; k < HID; ++k)
            s += (float)Abuf[1][m * KPAD + k] * Wc2[c * HID + k];
        out[(wg * BQ + m) * NCLS + c] = s;
    }
}

// ---------------------------------------------------------------------------
extern "C" void kernel_launch(void* const* d_in, const int* in_sizes, int n_in,
                              void* d_out, int out_size, void* d_ws, size_t ws_size,
                              hipStream_t stream)
{
    const float* coeffs = (const float*)d_in[0];
    const float* times  = (const float*)d_in[1];
    const float* W_init = (const float*)d_in[2];
    const float* b_init = (const float*)d_in[3];
    const float* W_in   = (const float*)d_in[4];
    const float* b_in   = (const float*)d_in[5];
    const float* W_h    = (const float*)d_in[6];
    const float* b_h    = (const float*)d_in[7];
    const float* W_out  = (const float*)d_in[8];
    const float* b_out  = (const float*)d_in[9];
    const float* Wc1    = (const float*)d_in[10];
    const float* bc1    = (const float*)d_in[11];
    const float* Wc2    = (const float*)d_in[12];
    const float* bc2    = (const float*)d_in[13];

    _Float16* Wf = (_Float16*)d_ws;                               // 5*256*288 halves
    float*    y0 = (float*)((char*)d_ws + (size_t)5 * WSZ * 2);   // 256*256 f32

    prep_weights<<<(5 * WSZ + 255) / 256, 256, 0, stream>>>(W_in, W_h, W_out, Wc1, Wf);
    init_y0<<<(BATCH * HID) / 256, 256, 0, stream>>>(coeffs, W_init, b_init, y0);
    cde_kernel<<<NWG, 256, 0, stream>>>(coeffs, times, Wf,
                                        b_in, b_h, b_out, bc1, Wc2, bc2,
                                        y0, (float*)d_out);
}
